// NodeModel_38285338477049
// MI455X (gfx1250) — compile-verified, matrix operand order checked by reading, not run on hardware
//
#include <hip/hip_runtime.h>

// ---------------------------------------------------------------------------
// Types for CDNA5 WMMA (wave32): D = A(16x32 bf16) * B(32x16 bf16) + C(16x16 f32)
// ---------------------------------------------------------------------------
typedef __bf16 bf16_t;
typedef __attribute__((ext_vector_type(16))) __bf16 v16bf;
typedef __attribute__((ext_vector_type(8)))  __bf16 v8bf;
typedef __attribute__((ext_vector_type(8)))  float  v8f;

#define NNODES 50000
#define NEDGES 800000
#define NFEAT  128
#define EFEAT  64
#define LSIZE  256
#define D1DIM  193      // NF + EF + 1
#define D2DIM  385      // NF + L + 1
#define K1PAD  224      // D1 padded to 7*32
#define K2PAD  416      // D2 padded to 13*32
#define C2PAD  400      // D2 cols padded to 25*16
#define KT1 7
#define CT1 16          // 256/16
#define KT2 13
#define CT2 25          // ceil(385/16)
#define KT3 13
#define CT3 8           // 128/16

// A-fragment loader from a row-major bf16 matrix (works for LDS or global via
// flat addressing). ISA 16-bit A 16x32 layout: lane l holds row (l&15);
// elements 0..7 = K[kb + hi*8 .. +7], elements 8..15 = K[kb+16+hi*8 .. +7],
// hi = l>>4. Both runs are contiguous -> two 16-byte loads.
__device__ __forceinline__ v16bf load_a_frag(const bf16_t* A, int ld, int rowBase,
                                             int kb, int lane) {
    int r  = rowBase + (lane & 15);
    int hi = lane >> 4;
    const bf16_t* p = A + (size_t)r * ld + kb + hi * 8;
    v8bf lo = *(const v8bf*)(p);
    v8bf hv = *(const v8bf*)(p + 16);
    return __builtin_shufflevector(lo, hv, 0,1,2,3,4,5,6,7,8,9,10,11,12,13,14,15);
}

__device__ __forceinline__ v16bf load_b_frag(const bf16_t* Bp, int kt, int ct,
                                             int ct_n, int lane) {
    return *(const v16bf*)(Bp + (((size_t)(kt * ct_n + ct) * 32 + lane) << 4));
}

// ---------------------------------------------------------------------------
// Zero a float region
// ---------------------------------------------------------------------------
__global__ void zero_kernel(float* p, long n) {
    long i = (long)blockIdx.x * blockDim.x + threadIdx.x;
    if (i < n) p[i] = 0.f;
}

// ---------------------------------------------------------------------------
// Pack a row-major f32 weight [rows x cols] into per-lane WMMA B-fragment
// order with zero K/N padding.  out[((kt*ct_n+ct)*32+lane)*16 + e]
// ---------------------------------------------------------------------------
__global__ void pack_kernel(const float* __restrict__ W, int rows, int cols,
                            int ct_n, int total, bf16_t* __restrict__ out) {
    int idx = blockIdx.x * blockDim.x + threadIdx.x;
    if (idx >= total) return;
    int e    = idx & 15;
    int lane = (idx >> 4) & 31;
    int rest = idx >> 9;
    int ct   = rest % ct_n;
    int kt   = rest / ct_n;
    int hi   = lane >> 4;
    int K    = kt * 32 + ((e >= 8) ? 16 : 0) + hi * 8 + (e & 7);
    int n    = ct * 16 + (lane & 15);
    float v  = (K < rows && n < cols) ? W[(size_t)K * cols + n] : 0.f;
    out[idx] = (bf16_t)v;
}

// ---------------------------------------------------------------------------
// Per-destination-node edge counts
// ---------------------------------------------------------------------------
__global__ void count_kernel(const int* __restrict__ eidx, float* __restrict__ cnt) {
    int e = blockIdx.x * blockDim.x + threadIdx.x;
    if (e < NEDGES) atomicAdd(&cnt[eidx[NEDGES + e]], 1.f);
}

// ---------------------------------------------------------------------------
// Edge GEMM. Block = 256 threads (8 waves), 32 edges x 256 cols per block.
// PASS 1: accumulate per-column sum/sumsq of (A@W1 + b1) for BatchNorm.
// PASS 2: recompute GEMM, apply BN (scale/shift) + ReLU, scatter-add to ssum.
// ---------------------------------------------------------------------------
template <int PASS>
__global__ void edge_kernel(const float* __restrict__ x,
                            const int* __restrict__ eidx,
                            const float* __restrict__ eattr,
                            const int* __restrict__ batch,
                            const bf16_t* __restrict__ W1p,
                            const float* __restrict__ b1,
                            const float* __restrict__ scale1,
                            const float* __restrict__ shift1,
                            float* __restrict__ colsum,
                            float* __restrict__ colsq,
                            float* __restrict__ ssum) {
    __shared__ __align__(16) bf16_t sA[32 * K1PAD];
    __shared__ int   sCol[32];
    __shared__ float sSum[LSIZE];
    __shared__ float sSq[LSIZE];

    const int t  = threadIdx.x;
    const int eb = blockIdx.x * 32;

    if (PASS == 1) { sSum[t] = 0.f; sSq[t] = 0.f; }
    if (t < 32) sCol[t] = eidx[NEDGES + eb + t];

    // Stage gathered+concatenated A rows (bf16) into LDS: 8 threads per edge.
    {
        int e   = t >> 3;
        int sub = t & 7;
        int row = eidx[eb + e];
        const float* xr = x + (size_t)row * NFEAT;
        const float* ea = eattr + (size_t)(eb + e) * EFEAT;
        float bc = (float)batch[row];
        bf16_t* dst = sA + e * K1PAD;
        #pragma unroll
        for (int j0 = 0; j0 < 28; ++j0) {
            int j = sub * 28 + j0;
            float v;
            if      (j < NFEAT)  v = xr[j];
            else if (j == NFEAT) v = bc;
            else if (j < D1DIM)  v = ea[j - NFEAT - 1];
            else                 v = 0.f;
            dst[j] = (bf16_t)v;
        }
    }
    __syncthreads();

    const int lane = t & 31;
    const int wave = t >> 5;
    const int rt   = (wave & 1) * 16;      // row tile within the 32 staged edges
    const int ctb  = (wave >> 1) * 4;      // 4 col tiles per wave

    v8f z = {0.f,0.f,0.f,0.f,0.f,0.f,0.f,0.f};
    v8f acc[4] = {z, z, z, z};

    for (int kt = 0; kt < KT1; ++kt) {
        v16bf af = load_a_frag(sA, K1PAD, rt, kt * 32, lane);
        #pragma unroll
        for (int i = 0; i < 4; ++i) {
            v16bf bfr = load_b_frag(W1p, kt, ctb + i, CT1, lane);
            acc[i] = __builtin_amdgcn_wmma_f32_16x16x32_bf16(
                false, af, false, bfr, (short)0, acc[i], false, false);
        }
    }

    const int hi = lane >> 4;
    #pragma unroll
    for (int i = 0; i < 4; ++i) {
        int n = (ctb + i) * 16 + (lane & 15);
        float bias = b1[n];
        if (PASS == 1) {
            float s = 0.f, sq = 0.f;
            #pragma unroll
            for (int v = 0; v < 8; ++v) {
                float val = acc[i][v] + bias;
                s += val; sq += val * val;
            }
            atomicAdd(&sSum[n], s);
            atomicAdd(&sSq[n], sq);
        } else {
            float sc = scale1[n], sh = shift1[n];
            #pragma unroll
            for (int v = 0; v < 8; ++v) {
                int m = v + 8 * hi;                  // edge row within tile
                float val = acc[i][v] + bias;
                float y = fmaxf(val * sc + sh, 0.f);
                int dest = sCol[rt + m];
                atomicAdd(&ssum[(size_t)dest * LSIZE + n], y);
            }
        }
    }

    if (PASS == 1) {
        __syncthreads();
        atomicAdd(&colsum[t], sSum[t]);
        atomicAdd(&colsq[t],  sSq[t]);
    }
}

// ---------------------------------------------------------------------------
// Fold BN stats into scale/shift: scale = g*rsqrt(var+eps), shift = be - mu*scale
// ---------------------------------------------------------------------------
__global__ void finalize_stats(const float* __restrict__ sum,
                               const float* __restrict__ sq,
                               const float* __restrict__ g,
                               const float* __restrict__ be,
                               float inv_count, int ncols,
                               float* __restrict__ scale,
                               float* __restrict__ shift) {
    int c = blockIdx.x * blockDim.x + threadIdx.x;
    if (c >= ncols) return;
    float mu  = sum[c] * inv_count;
    float var = sq[c] * inv_count - mu * mu;
    float rs  = rsqrtf(var + 1e-5f);
    float sc  = g[c] * rs;
    scale[c] = sc;
    shift[c] = be[c] - mu * sc;
}

// ---------------------------------------------------------------------------
// Build A2 = [x | batch | ssum/max(cnt,1) | 0 pad]  -> bf16 [N x 416]
// ---------------------------------------------------------------------------
__global__ void build_a2(const float* __restrict__ x,
                         const int* __restrict__ batch,
                         const float* __restrict__ ssum,
                         const float* __restrict__ cnt,
                         bf16_t* __restrict__ a2) {
    long idx = (long)blockIdx.x * blockDim.x + threadIdx.x;
    if (idx >= (long)NNODES * K2PAD) return;
    int i = (int)(idx / K2PAD), j = (int)(idx % K2PAD);
    float v;
    if      (j < NFEAT)  v = x[(size_t)i * NFEAT + j];
    else if (j == NFEAT) v = (float)batch[i];
    else if (j < D2DIM)  v = ssum[(size_t)i * LSIZE + (j - NFEAT - 1)] / fmaxf(cnt[i], 1.f);
    else                 v = 0.f;
    a2[idx] = (bf16_t)v;
}

// ---------------------------------------------------------------------------
// Node GEMM1: h2pre = A2 @ W2 + b2 (bf16 store, [N x 400]) + BN column stats
// grid (3125 row-tiles, 7), block 128 = 4 waves, wave col-tile = y*4+w (<25)
// ---------------------------------------------------------------------------
__global__ void node_gemm1(const bf16_t* __restrict__ a2,
                           const bf16_t* __restrict__ W2p,
                           const float* __restrict__ b2,
                           bf16_t* __restrict__ h2pre,
                           float* __restrict__ colsum2,
                           float* __restrict__ colsq2) {
    __shared__ float sSum[64], sSq[64];
    const int t = threadIdx.x, lane = t & 31, wave = t >> 5;
    const int rb = blockIdx.x * 16;
    const int ct = blockIdx.y * 4 + wave;
    if (t < 64) { sSum[t] = 0.f; sSq[t] = 0.f; }
    __syncthreads();

    if (ct < CT2) {
        v8f acc = {0.f,0.f,0.f,0.f,0.f,0.f,0.f,0.f};
        for (int kt = 0; kt < KT2; ++kt) {
            v16bf af  = load_a_frag(a2, K2PAD, rb, kt * 32, lane);
            v16bf bfr = load_b_frag(W2p, kt, ct, CT2, lane);
            acc = __builtin_amdgcn_wmma_f32_16x16x32_bf16(
                false, af, false, bfr, (short)0, acc, false, false);
        }
        int n  = ct * 16 + (lane & 15);
        int hi = lane >> 4;
        float bias = (n < D2DIM) ? b2[n] : 0.f;
        float s = 0.f, sq = 0.f;
        #pragma unroll
        for (int v = 0; v < 8; ++v) {
            float val = acc[v] + bias;
            h2pre[(size_t)(rb + v + 8 * hi) * C2PAD + n] = (bf16_t)val;
            s += val; sq += val * val;
        }
        if (n < D2DIM) {
            int li = wave * 16 + (lane & 15);
            atomicAdd(&sSum[li], s);
            atomicAdd(&sSq[li], sq);
        }
    }
    __syncthreads();
    if (t < 64) {
        int c = blockIdx.y * 64 + t;
        if (c < D2DIM) {
            atomicAdd(&colsum2[c], sSum[t]);
            atomicAdd(&colsq2[c],  sSq[t]);
        }
    }
}

// ---------------------------------------------------------------------------
// Build A3 = ReLU(BN(h2pre)) -> bf16 [N x 416] (zero K-pad)
// ---------------------------------------------------------------------------
__global__ void build_a3(const bf16_t* __restrict__ h2pre,
                         const float* __restrict__ scale2,
                         const float* __restrict__ shift2,
                         bf16_t* __restrict__ a3) {
    long idx = (long)blockIdx.x * blockDim.x + threadIdx.x;
    if (idx >= (long)NNODES * K2PAD) return;
    int i = (int)(idx / K2PAD), j = (int)(idx % K2PAD);
    float v = 0.f;
    if (j < D2DIM) {
        float h = (float)h2pre[(size_t)i * C2PAD + j];
        v = fmaxf(h * scale2[j] + shift2[j], 0.f);
    }
    a3[idx] = (bf16_t)v;
}

// ---------------------------------------------------------------------------
// Node GEMM2: out = ReLU(A3 @ W3 + b3)  [N x 128] f32
// grid (3125, 2), block 128 = 4 waves, wave col-tile = y*4+w (8 tiles exact)
// ---------------------------------------------------------------------------
__global__ void node_gemm2(const bf16_t* __restrict__ a3,
                           const bf16_t* __restrict__ W3p,
                           const float* __restrict__ b3,
                           float* __restrict__ out) {
    const int t = threadIdx.x, lane = t & 31, wave = t >> 5;
    const int rb = blockIdx.x * 16;
    const int ct = blockIdx.y * 4 + wave;

    v8f acc = {0.f,0.f,0.f,0.f,0.f,0.f,0.f,0.f};
    for (int kt = 0; kt < KT3; ++kt) {
        v16bf af  = load_a_frag(a3, K2PAD, rb, kt * 32, lane);
        v16bf bfr = load_b_frag(W3p, kt, ct, CT3, lane);
        acc = __builtin_amdgcn_wmma_f32_16x16x32_bf16(
            false, af, false, bfr, (short)0, acc, false, false);
    }
    int n  = ct * 16 + (lane & 15);
    int hi = lane >> 4;
    float bias = b3[n];
    #pragma unroll
    for (int v = 0; v < 8; ++v) {
        out[(size_t)(rb + v + 8 * hi) * NFEAT + n] = fmaxf(acc[v] + bias, 0.f);
    }
}

// ---------------------------------------------------------------------------
// Launch
// ---------------------------------------------------------------------------
extern "C" void kernel_launch(void* const* d_in, const int* in_sizes, int n_in,
                              void* d_out, int out_size, void* d_ws, size_t ws_size,
                              hipStream_t stream) {
    const float* x     = (const float*)d_in[0];
    const int*   eidx  = (const int*)  d_in[1];
    const float* eattr = (const float*)d_in[2];
    // d_in[3] = u (unused by reference)
    const int*   batch = (const int*)  d_in[4];
    const float* W1    = (const float*)d_in[5];
    const float* b1    = (const float*)d_in[6];
    const float* g1    = (const float*)d_in[7];
    const float* be1   = (const float*)d_in[8];
    const float* W2    = (const float*)d_in[9];
    const float* b2    = (const float*)d_in[10];
    const float* g2    = (const float*)d_in[11];
    const float* be2   = (const float*)d_in[12];
    const float* W3    = (const float*)d_in[13];
    const float* b3    = (const float*)d_in[14];
    float* out = (float*)d_out;

    char* ws = (char*)d_ws;
    size_t off = 0;
    auto take = [&](size_t bytes) -> char* {
        char* p = ws + off;
        off = (off + bytes + 255) & ~(size_t)255;
        return p;
    };
    // zero-initialized region (contiguous at start of ws)
    float* ssum    = (float*)take((size_t)NNODES * LSIZE * 4);
    float* cnt     = (float*)take((size_t)NNODES * 4);
    float* colsum1 = (float*)take(256 * 4);
    float* colsq1  = (float*)take(256 * 4);
    float* colsum2 = (float*)take(C2PAD * 4);
    float* colsq2  = (float*)take(C2PAD * 4);
    long zero_floats = (long)(off / 4);
    // remaining scratch
    float*  scale1 = (float*)take(256 * 4);
    float*  shift1 = (float*)take(256 * 4);
    float*  scale2 = (float*)take(C2PAD * 4);
    float*  shift2 = (float*)take(C2PAD * 4);
    bf16_t* W1p    = (bf16_t*)take((size_t)KT1 * CT1 * 512 * 2);
    bf16_t* W2p    = (bf16_t*)take((size_t)KT2 * CT2 * 512 * 2);
    bf16_t* W3p    = (bf16_t*)take((size_t)KT3 * CT3 * 512 * 2);
    bf16_t* a2     = (bf16_t*)take((size_t)NNODES * K2PAD * 2);
    bf16_t* h2pre  = (bf16_t*)take((size_t)NNODES * C2PAD * 2);
    bf16_t* a3     = (bf16_t*)take((size_t)NNODES * K2PAD * 2);

    // 0. zero accumulators
    zero_kernel<<<(unsigned)((zero_floats + 255) / 256), 256, 0, stream>>>(
        (float*)ws, zero_floats);

    // 1. pack weights into WMMA B-fragment order (bf16, zero padded)
    {
        int t1 = KT1 * CT1 * 512;
        pack_kernel<<<(t1 + 255) / 256, 256, 0, stream>>>(W1, D1DIM, LSIZE, CT1, t1, W1p);
        int t2 = KT2 * CT2 * 512;
        pack_kernel<<<(t2 + 255) / 256, 256, 0, stream>>>(W2, D2DIM, D2DIM, CT2, t2, W2p);
        int t3 = KT3 * CT3 * 512;
        pack_kernel<<<(t3 + 255) / 256, 256, 0, stream>>>(W3, D2DIM, NFEAT, CT3, t3, W3p);
    }

    // 2. per-node edge counts
    count_kernel<<<NEDGES / 256, 256, 0, stream>>>(eidx, cnt);

    // 3. edge GEMM pass 1: BN statistics
    edge_kernel<1><<<NEDGES / 32, 256, 0, stream>>>(
        x, eidx, eattr, batch, W1p, b1, nullptr, nullptr, colsum1, colsq1, nullptr);

    // 4. fold stats
    finalize_stats<<<1, 256, 0, stream>>>(colsum1, colsq1, g1, be1,
                                          1.0f / (float)NEDGES, LSIZE, scale1, shift1);

    // 5. edge GEMM pass 2: BN + ReLU + scatter-add
    edge_kernel<2><<<NEDGES / 32, 256, 0, stream>>>(
        x, eidx, eattr, batch, W1p, b1, scale1, shift1, nullptr, nullptr, ssum);

    // 6. build node-MLP input
    {
        long tot = (long)NNODES * K2PAD;
        build_a2<<<(unsigned)((tot + 255) / 256), 256, 0, stream>>>(x, batch, ssum, cnt, a2);
    }

    // 7. node GEMM1 + BN stats
    node_gemm1<<<dim3(NNODES / 16, 7), 128, 0, stream>>>(a2, W2p, b2, h2pre, colsum2, colsq2);

    // 8. fold stats
    finalize_stats<<<2, 256, 0, stream>>>(colsum2, colsq2, g2, be2,
                                          1.0f / (float)NNODES, D2DIM, scale2, shift2);

    // 9. BN + ReLU -> A3
    {
        long tot = (long)NNODES * K2PAD;
        build_a3<<<(unsigned)((tot + 255) / 256), 256, 0, stream>>>(h2pre, scale2, shift2, a3);
    }

    // 10. node GEMM2 -> output
    node_gemm2<<<dim3(NNODES / 16, 2), 128, 0, stream>>>(a3, W3p, b3, out);
}
